// HandRefiner_66846870995359
// MI455X (gfx1250) — matrix operand order repaired and checked by reading, not compiled, not run on hardware
//
#include <hip/hip_runtime.h>
#include <hip/hip_bf16.h>
#include <math.h>

typedef __bf16 bf16;
typedef __bf16 v16bf __attribute__((ext_vector_type(16)));
typedef float  v8f   __attribute__((ext_vector_type(8)));

#define BB 32
#define LL 128
#define NN 512
#define JJN 21
#define HDIM 99
#define DD 512
#define HH 8
#define DH 64
#define DFFN 2048
#define CATR 1249
#define CATP 1280
#define TOK 256            // 2*L tokens per batch
#define ROWS 8192          // B*TOK
#define HROWS 4096         // B*L
#define HPAD 128           // padded output-head N

// ---------------------------------------------------------------------------
// CDNA5 async global->LDS copy (16B), tracked by ASYNCcnt.
// ---------------------------------------------------------------------------
__device__ __forceinline__ void async_copy16(void* lds, const void* gsrc) {
    unsigned l = (unsigned)(unsigned long long)lds;   // low 32 bits = LDS byte address
    asm volatile("global_load_async_to_lds_b128 %0, %1, off"
                 :: "v"(l), "v"(gsrc) : "memory");
}
__device__ __forceinline__ void async_wait0() {
    asm volatile("s_wait_asynccnt 0" ::: "memory");
}

// ---------------------------------------------------------------------------
// Weight prep: f32 [K,N] -> bf16 transposed+padded [Npad, Kpad] (zero-filled).
// Makes every GEMM tile load dense/contiguous (no bounds checks in hot loop).
// ---------------------------------------------------------------------------
__global__ void convert_T_kernel(const float* __restrict__ src, bf16* __restrict__ dst,
                                 int K, int N, int Kpad, int Npad) {
    long i = (long)blockIdx.x * 256 + threadIdx.x;
    long tot = (long)Npad * Kpad;
    if (i >= tot) return;
    int n = (int)(i / Kpad);
    int k = (int)(i % Kpad);
    float v = (k < K && n < N) ? src[(long)k * N + n] : 0.0f;
    dst[i] = (bf16)v;
}

// ---------------------------------------------------------------------------
// Feature assembly: [x_hand | j | m_contact | |pc| | att] -> bf16 [8192, 1280]
// one block per (hand, b, l); includes argmin nearest-point attention maps.
// ---------------------------------------------------------------------------
__global__ __launch_bounds__(256) void features_kernel(
    const float* __restrict__ xl, const float* __restrict__ xr,
    const float* __restrict__ jl, const float* __restrict__ jr,
    const float* __restrict__ mc, const float* __restrict__ pc,
    bf16* __restrict__ feat)
{
    int g = blockIdx.x;            // 0..8191
    int hand = g >> 12;            // 0 = left, 1 = right
    int bl = g & 4095;             // b*L + l
    int b = bl >> 7;
    int t = threadIdx.x;

    __shared__ float spc[NN * 3];  // 6 KB point cloud for this (b,l)
    __shared__ float sj[64];
    __shared__ float rd[256];
    __shared__ int   ri[256];
    __shared__ float snear[3];

    const float* xh = hand ? xr : xl;
    const float* jh = hand ? jr : jl;

    long pcb = (long)bl * NN * 3;
    for (int i = t; i < NN * 3; i += 256) spc[i] = pc[pcb + i];
    if (t < 63) sj[t] = jh[(long)bl * 63 + t];
    __syncthreads();

    bf16* fr = feat + (long)g * CATP;
    if (t < HDIM) fr[t] = (bf16)xh[(long)bl * HDIM + t];
    if (t < 63)   fr[HDIM + t] = (bf16)sj[t];
    for (int p = t; p < NN; p += 256) {
        fr[162 + p] = (bf16)mc[(long)b * NN + p];
        float px = spc[p * 3], py = spc[p * 3 + 1], pz = spc[p * 3 + 2];
        fr[674 + p] = (bf16)sqrtf(px * px + py * py + pz * pz);
    }
    if (t < (CATP - CATR)) fr[CATR + t] = (bf16)0.0f;   // zero K-padding

    for (int jj = 0; jj < JJN; ++jj) {
        float jx = sj[jj * 3], jy = sj[jj * 3 + 1], jz = sj[jj * 3 + 2];
        float best = 3.4e38f; int bi = 0;
        for (int p = t; p < NN; p += 256) {
            float dx = jx - spc[p * 3];
            float dy = jy - spc[p * 3 + 1];
            float dz = jz - spc[p * 3 + 2];
            float d2 = dx * dx + dy * dy + dz * dz;
            if (d2 < best) { best = d2; bi = p; }
        }
        rd[t] = best; ri[t] = bi;
        __syncthreads();
        for (int s = 128; s > 0; s >>= 1) {
            if (t < s) {
                if (rd[t + s] < rd[t] || (rd[t + s] == rd[t] && ri[t + s] < ri[t])) {
                    rd[t] = rd[t + s]; ri[t] = ri[t + s];
                }
            }
            __syncthreads();
        }
        if (t < 3) snear[t] = spc[ri[0] * 3 + t];
        __syncthreads();
        if (t < 3) {
            float dj = sj[jj * 3 + t] - snear[t];
            fr[1186 + jj * 3 + t] = (bf16)expf(-50.0f * dj * dj);
        }
        __syncthreads();
    }
}

// ---------------------------------------------------------------------------
// bf16 WMMA GEMM: C[M,N] = A[M,K] * Bt[N,K]^T (+bias, +relu).
// A row-major [M,K] (M multiple of 64, K multiple of 32, both dense);
// Bt pre-transposed/padded [Npad,K] so tiles are dense contiguous rows.
// Double-buffered async global->LDS staging (ASYNCcnt) overlapped with WMMA;
// one barrier per K-step. 128 thr = 4 waves, tile 64x64x32, per-wave 2x2 WMMA.
// Epilogue behavior selected at compile time.
// ---------------------------------------------------------------------------
template<bool HASBIAS, bool RELU, bool HASCF, bool HASCB, bool STORET>
__global__ __launch_bounds__(128) void gemm_bf16_kernel(
    const bf16* __restrict__ A, int lda,
    const bf16* __restrict__ Bt, int ldbt,
    float* __restrict__ Cf, int ldc,
    bf16* __restrict__ Cb, long ldcb,
    const float* __restrict__ bias,
    int N, int K)
{
    __shared__ alignas(64) bf16 sA[2][64 * 32];   // [m][k]
    __shared__ alignas(64) bf16 sB[2][64 * 32];   // [n][k]

    int bm = blockIdx.x * 64;
    int bn = blockIdx.y * 64;
    int t = threadIdx.x;
    int wave = t >> 5, lane = t & 31;
    int wm = (wave & 1) * 32, wn = (wave >> 1) * 32;
    int ln15 = lane & 15, hi = lane >> 4;

    v8f acc[2][2];
    for (int i = 0; i < 2; ++i)
        for (int j = 0; j < 2; ++j)
            for (int r = 0; r < 8; ++r) acc[i][j][r] = 0.0f;

    int ar = t >> 2;           // 0..31
    int ac = (t & 3) * 8;      // 0,8,16,24 (bf16 elements; 16B chunk)

    auto stage = [&](int buf, int k0) {
        for (int p = 0; p < 2; ++p) {
            int row = p * 32 + ar;
            async_copy16(&sA[buf][row * 32 + ac], A  + (long)(bm + row) * lda  + k0 + ac);
            async_copy16(&sB[buf][row * 32 + ac], Bt + (long)(bn + row) * ldbt + k0 + ac);
        }
    };

    stage(0, 0);               // prologue: tile 0 in flight
    int cur = 0;
    for (int k0 = 0; k0 < K; k0 += 32) {
        async_wait0();         // this wave's copies for buf[cur] landed
        __syncthreads();       // everyone's landed; buf[cur^1] free to overwrite
        if (k0 + 32 < K) stage(cur ^ 1, k0 + 32);   // prefetch under the math

        const bf16* pA = sA[cur];
        const bf16* pB = sB[cur];
        v16bf af[2], bfv[2];
        for (int i = 0; i < 2; ++i) {
            const bf16* ap = pA + (wm + i * 16 + ln15) * 32 + hi * 16;
            ((uint4*)&af[i])[0] = *(const uint4*)(ap);
            ((uint4*)&af[i])[1] = *(const uint4*)(ap + 8);
        }
        for (int j = 0; j < 2; ++j) {
            const bf16* bp = pB + (wn + j * 16 + ln15) * 32 + hi * 16;
            bfv[j] = *(const v16bf*)bp;
        }
        for (int i = 0; i < 2; ++i)
            for (int j = 0; j < 2; ++j)
                acc[i][j] = __builtin_amdgcn_wmma_f32_16x16x32_bf16(
                    false, af[i], false, bfv[j], (short)0, acc[i][j], false, false);
        cur ^= 1;
    }

    for (int i = 0; i < 2; ++i)
        for (int j = 0; j < 2; ++j)
            for (int r = 0; r < 8; ++r) {
                int row = bm + wm + i * 16 + hi * 8 + r;
                int col = bn + wn + j * 16 + ln15;
                if (col < N) {
                    float v = acc[i][j][r];
                    if (HASBIAS) v += bias[col];
                    if (RELU) v = fmaxf(v, 0.0f);
                    if (HASCF) Cf[(long)row * ldc + col] = v;
                    if (HASCB) {
                        if (STORET) Cb[(long)col * ldcb + row] = (bf16)v;
                        else        Cb[(long)row * ldcb + col] = (bf16)v;
                    }
                }
            }
}

// ---------------------------------------------------------------------------
// Positional encodings (frame + agent) + masks; updates x f32 in-place and
// writes bf16 mirror.
// ---------------------------------------------------------------------------
__global__ void pe_mask_kernel(float* __restrict__ x, bf16* __restrict__ xb,
                               const float* __restrict__ ml, const float* __restrict__ mr,
                               const unsigned char* __restrict__ fpm)
{
    long i = (long)blockIdx.x * 256 + threadIdx.x;
    if (i >= (long)ROWS * DD) return;
    int d = (int)(i & (DD - 1));
    int row = (int)(i >> 9);         // b*256 + t
    int b = row >> 8;
    int tt = row & 255;
    int ii = d >> 1;
    float freq = expf(-logf(10000.0f) * (2.0f * (float)ii / (float)DD));
    float af = (float)(tt >> 1) * freq;
    float aa = (float)(tt & 1) * freq;
    float pe = (d & 1) ? (cosf(af) + cosf(aa)) : (sinf(af) + sinf(aa));
    float m = ((tt & 1) ? mr : ml)[b * LL + (tt >> 1)];
    float f = fpm[row] ? 0.0f : 1.0f;
    float v = (x[i] + pe) * m * f;
    x[i] = v;
    xb[i] = (bf16)v;
}

// ---------------------------------------------------------------------------
// Fused attention per (b, h, 32-query tile): scores WMMA -> softmax -> PV WMMA.
// 64 threads = 2 waves, each wave owns 16 query rows. V is pre-transposed
// ([D, ROWS]) by its projection GEMM so PV B-fragments load contiguously.
// ---------------------------------------------------------------------------
__global__ __launch_bounds__(64) void attention_kernel(
    const bf16* __restrict__ Q, const bf16* __restrict__ Kt,
    const bf16* __restrict__ VT, const unsigned char* __restrict__ fpm,
    bf16* __restrict__ O)
{
    int g = blockIdx.x;                 // B*H*8 = 2048
    int q0 = (g & 7) * 32;
    int h = (g >> 3) & 7;
    int b = g >> 6;
    int wave = threadIdx.x >> 5, lane = threadIdx.x & 31;
    int ln15 = lane & 15, hi = lane >> 4;

    __shared__ alignas(64) float sS[32 * TOK];      // scores (f32)
    __shared__ alignas(64) bf16  sP[32 * TOK];      // probabilities (bf16)

    // ---- scores: S = Q K^T / sqrt(dh) + bias ----
    long qrowbase = (long)(b * TOK + q0 + wave * 16 + ln15) * DD + h * DH;
    v16bf aq[2];
    for (int ks = 0; ks < 2; ++ks) {
        const bf16* ap = Q + qrowbase + ks * 32 + hi * 16;
        ((uint4*)&aq[ks])[0] = *(const uint4*)(ap);
        ((uint4*)&aq[ks])[1] = *(const uint4*)(ap + 8);
    }
    for (int nt = 0; nt < TOK / 16; ++nt) {
        int n0 = nt * 16;
        v8f acc;
        for (int r = 0; r < 8; ++r) acc[r] = 0.0f;
        for (int ks = 0; ks < 2; ++ks) {
            // K row (time) contiguous over channels -> direct B-fragment load
            const bf16* bp = Kt + (long)(b * TOK + n0 + ln15) * DD + h * DH + ks * 32 + hi * 16;
            v16bf bk = *(const v16bf*)bp;
            acc = __builtin_amdgcn_wmma_f32_16x16x32_bf16(
                false, aq[ks], false, bk, (short)0, acc, false, false);
        }
        int key = n0 + ln15;
        float kb = fpm[b * TOK + key] ? -1e9f : 0.0f;
        for (int r = 0; r < 8; ++r) {
            int qr = wave * 16 + hi * 8 + r;
            sS[qr * TOK + key] = acc[r] * 0.125f + kb;   // 1/sqrt(64)
        }
    }
    __syncthreads();

    // ---- softmax over keys (one thread per query row) ----
    if (threadIdx.x < 32) {
        float* rowp = sS + threadIdx.x * TOK;
        float mx = -1e30f;
        for (int i = 0; i < TOK; ++i) mx = fmaxf(mx, rowp[i]);
        float sum = 0.0f;
        for (int i = 0; i < TOK; ++i) { float e = expf(rowp[i] - mx); rowp[i] = e; sum += e; }
        float inv = 1.0f / sum;
        bf16* prow = sP + threadIdx.x * TOK;
        for (int i = 0; i < TOK; ++i) prow[i] = (bf16)(rowp[i] * inv);
    }
    __syncthreads();

    // ---- O = P V ---- (V transposed: VT[ch][b*TOK + t])
    v8f oacc[4];
    for (int c = 0; c < 4; ++c)
        for (int r = 0; r < 8; ++r) oacc[c][r] = 0.0f;

    for (int kt = 0; kt < TOK; kt += 32) {
        const bf16* pp = sP + (wave * 16 + ln15) * TOK + kt + hi * 16;
        v16bf ap;
        ((uint4*)&ap)[0] = *(const uint4*)(pp);
        ((uint4*)&ap)[1] = *(const uint4*)(pp + 8);
        for (int c = 0; c < 4; ++c) {
            const bf16* bp = VT + (long)(h * DH + c * 16 + ln15) * ROWS + b * TOK + kt + hi * 16;
            v16bf bv = *(const v16bf*)bp;
            oacc[c] = __builtin_amdgcn_wmma_f32_16x16x32_bf16(
                false, ap, false, bv, (short)0, oacc[c], false, false);
        }
    }

    for (int c = 0; c < 4; ++c)
        for (int r = 0; r < 8; ++r) {
            int q = q0 + wave * 16 + hi * 8 + r;
            int col = h * DH + c * 16 + ln15;
            O[(long)(b * TOK + q) * DD + col] = (bf16)oacc[c][r];
        }
}

// ---------------------------------------------------------------------------
// In-place residual + LayerNorm over D=512: x = LN(x + y); refresh bf16 mirror.
// ---------------------------------------------------------------------------
__global__ __launch_bounds__(256) void ln_kernel(float* __restrict__ x, const float* __restrict__ y,
                                                 bf16* __restrict__ xb,
                                                 const float* __restrict__ g, const float* __restrict__ bb)
{
    int row = blockIdx.x;
    int t = threadIdx.x;
    __shared__ float s1[256], s2[256];
    long base = (long)row * DD;
    float v0 = x[base + t] + y[base + t];
    float v1 = x[base + t + 256] + y[base + t + 256];
    s1[t] = v0 + v1;
    s2[t] = v0 * v0 + v1 * v1;
    __syncthreads();
    for (int s = 128; s > 0; s >>= 1) {
        if (t < s) { s1[t] += s1[t + s]; s2[t] += s2[t + s]; }
        __syncthreads();
    }
    float mean = s1[0] * (1.0f / DD);
    float var = s2[0] * (1.0f / DD) - mean * mean;
    float rstd = rsqrtf(var + 1e-5f);
    float o0 = (v0 - mean) * rstd * g[t] + bb[t];
    float o1 = (v1 - mean) * rstd * g[t + 256] + bb[t + 256];
    x[base + t] = o0;       x[base + t + 256] = o1;
    xb[base + t] = (bf16)o0; xb[base + t + 256] = (bf16)o1;
}

// ---------------------------------------------------------------------------
// Final output masking: out *= mask_hand * fpm
// ---------------------------------------------------------------------------
__global__ void mask_out_kernel(float* __restrict__ out,
                                const float* __restrict__ ml, const float* __restrict__ mr,
                                const unsigned char* __restrict__ fpm)
{
    int i = blockIdx.x * 256 + threadIdx.x;
    const int HALFSZ = HROWS * HDIM;   // 405504
    if (i >= 2 * HALFSZ) return;
    int half = i / HALFSZ;
    int rem = i - half * HALFSZ;
    int row = rem / HDIM;              // b*L + l
    int b = row >> 7, l = row & 127;
    int tt = 2 * l + half;
    float m = (half ? mr : ml)[row];
    float f = fpm[b * TOK + tt] ? 0.0f : 1.0f;
    out[i] *= m * f;
}

// ---------------------------------------------------------------------------
// Host-side orchestration
// ---------------------------------------------------------------------------
extern "C" void kernel_launch(void* const* d_in, const int* in_sizes, int n_in,
                              void* d_out, int out_size, void* d_ws, size_t ws_size,
                              hipStream_t stream)
{
    (void)in_sizes; (void)n_in; (void)out_size; (void)ws_size;

    const float* x_lhand = (const float*)d_in[0];
    const float* x_rhand = (const float*)d_in[1];
    const float* j_lhand = (const float*)d_in[2];
    const float* j_rhand = (const float*)d_in[3];
    const float* m_contact = (const float*)d_in[4];
    const float* ref_pc = (const float*)d_in[5];
    const float* mask_l = (const float*)d_in[6];
    const float* mask_r = (const float*)d_in[7];
    const unsigned char* fpm = (const unsigned char*)d_in[8];
    const float* W_fc_l = (const float*)d_in[9];
    const float* b_fc_l = (const float*)d_in[10];
    const float* W_fc_r = (const float*)d_in[11];
    const float* b_fc_r = (const float*)d_in[12];
    const float* Wq = (const float*)d_in[13];
    const float* Wk = (const float*)d_in[14];
    const float* Wv = (const float*)d_in[15];
    const float* Wo = (const float*)d_in[16];
    const float* W1 = (const float*)d_in[17];
    const float* b1 = (const float*)d_in[18];
    const float* W2 = (const float*)d_in[19];
    const float* b2 = (const float*)d_in[20];
    const float* ln1_g = (const float*)d_in[21];
    const float* ln1_b = (const float*)d_in[22];
    const float* ln2_g = (const float*)d_in[23];
    const float* ln2_b = (const float*)d_in[24];
    const float* W_out_l = (const float*)d_in[25];
    const float* b_out_l = (const float*)d_in[26];
    const float* W_out_r = (const float*)d_in[27];
    const float* b_out_r = (const float*)d_in[28];

    char* ws = (char*)d_ws;
    size_t off = 0;
    auto alloc = [&](size_t bytes) -> void* {
        off = (off + 255) & ~(size_t)255;
        void* p = ws + off;
        off += bytes;
        return p;
    };

    // all weights stored transposed+padded [Npad, Kpad] bf16
    bf16* wfcl = (bf16*)alloc((size_t)DD * CATP * 2);
    bf16* wfcr = (bf16*)alloc((size_t)DD * CATP * 2);
    bf16* wq   = (bf16*)alloc((size_t)DD * DD * 2);
    bf16* wk   = (bf16*)alloc((size_t)DD * DD * 2);
    bf16* wv   = (bf16*)alloc((size_t)DD * DD * 2);
    bf16* wo   = (bf16*)alloc((size_t)DD * DD * 2);
    bf16* w1   = (bf16*)alloc((size_t)DFFN * DD * 2);
    bf16* w2   = (bf16*)alloc((size_t)DD * DFFN * 2);
    bf16* wol  = (bf16*)alloc((size_t)HPAD * DD * 2);
    bf16* wor  = (bf16*)alloc((size_t)HPAD * DD * 2);
    bf16* feat = (bf16*)alloc((size_t)ROWS * CATP * 2);
    float* x   = (float*)alloc((size_t)ROWS * DD * 4);
    bf16* xb   = (bf16*)alloc((size_t)ROWS * DD * 2);
    bf16* qb   = (bf16*)alloc((size_t)ROWS * DD * 2);
    bf16* kb   = (bf16*)alloc((size_t)ROWS * DD * 2);
    bf16* vbT  = (bf16*)alloc((size_t)DD * ROWS * 2);   // transposed V
    bf16* ob   = (bf16*)alloc((size_t)ROWS * DD * 2);
    float* y   = (float*)alloc((size_t)ROWS * DD * 4);
    bf16* hbuf = (bf16*)alloc((size_t)ROWS * DFFN * 2);

    auto cgrid = [](long n) { return (unsigned)((n + 255) / 256); };

    // --- weight conversions: transpose + pad to dense WMMA-friendly tiles ---
    convert_T_kernel<<<cgrid((long)DD * CATP), 256, 0, stream>>>(W_fc_l, wfcl, CATR, DD, CATP, DD);
    convert_T_kernel<<<cgrid((long)DD * CATP), 256, 0, stream>>>(W_fc_r, wfcr, CATR, DD, CATP, DD);
    convert_T_kernel<<<cgrid((long)DD * DD), 256, 0, stream>>>(Wq, wq, DD, DD, DD, DD);
    convert_T_kernel<<<cgrid((long)DD * DD), 256, 0, stream>>>(Wk, wk, DD, DD, DD, DD);
    convert_T_kernel<<<cgrid((long)DD * DD), 256, 0, stream>>>(Wv, wv, DD, DD, DD, DD);
    convert_T_kernel<<<cgrid((long)DD * DD), 256, 0, stream>>>(Wo, wo, DD, DD, DD, DD);
    convert_T_kernel<<<cgrid((long)DFFN * DD), 256, 0, stream>>>(W1, w1, DD, DFFN, DD, DFFN);
    convert_T_kernel<<<cgrid((long)DD * DFFN), 256, 0, stream>>>(W2, w2, DFFN, DD, DFFN, DD);
    convert_T_kernel<<<cgrid((long)HPAD * DD), 256, 0, stream>>>(W_out_l, wol, DD, HDIM, DD, HPAD);
    convert_T_kernel<<<cgrid((long)HPAD * DD), 256, 0, stream>>>(W_out_r, wor, DD, HDIM, DD, HPAD);

    // --- feature assembly (both hands) ---
    features_kernel<<<ROWS, 256, 0, stream>>>(x_lhand, x_rhand, j_lhand, j_rhand,
                                              m_contact, ref_pc, feat);

    // --- input projections; interleave left/right into x[B,256,D] via ldc=1024 ---
    gemm_bf16_kernel<true, false, true, false, false><<<dim3(HROWS / 64, DD / 64), 128, 0, stream>>>(
        feat, CATP, wfcl, CATP, x, 2 * DD, nullptr, 0, b_fc_l, DD, CATP);
    gemm_bf16_kernel<true, false, true, false, false><<<dim3(HROWS / 64, DD / 64), 128, 0, stream>>>(
        feat + (size_t)HROWS * CATP, CATP, wfcr, CATP, x + DD, 2 * DD, nullptr, 0, b_fc_r, DD, CATP);

    // --- positional encodings + masks; build bf16 mirror ---
    pe_mask_kernel<<<cgrid((long)ROWS * DD), 256, 0, stream>>>(x, xb, mask_l, mask_r, fpm);

    // --- Q, K projections (bf16) and V projection stored transposed ---
    gemm_bf16_kernel<false, false, false, true, false><<<dim3(ROWS / 64, DD / 64), 128, 0, stream>>>(
        xb, DD, wq, DD, nullptr, 0, qb, DD, nullptr, DD, DD);
    gemm_bf16_kernel<false, false, false, true, false><<<dim3(ROWS / 64, DD / 64), 128, 0, stream>>>(
        xb, DD, wk, DD, nullptr, 0, kb, DD, nullptr, DD, DD);
    gemm_bf16_kernel<false, false, false, true, true><<<dim3(ROWS / 64, DD / 64), 128, 0, stream>>>(
        xb, DD, wv, DD, nullptr, 0, vbT, (long)ROWS, nullptr, DD, DD);

    // --- fused attention ---
    attention_kernel<<<BB * HH * (TOK / 32), 64, 0, stream>>>(qb, kb, vbT, fpm, ob);

    // --- output projection Wo -> y (f32) ---
    gemm_bf16_kernel<false, false, true, false, false><<<dim3(ROWS / 64, DD / 64), 128, 0, stream>>>(
        ob, DD, wo, DD, y, DD, nullptr, 0, nullptr, DD, DD);

    // --- residual + LN1 (in place on x; refresh xb) ---
    ln_kernel<<<ROWS, 256, 0, stream>>>(x, y, xb, ln1_g, ln1_b);

    // --- FFN: relu(x W1 + b1) -> h (bf16);  h W2 + b2 -> y (f32) ---
    gemm_bf16_kernel<true, true, false, true, false><<<dim3(ROWS / 64, DFFN / 64), 128, 0, stream>>>(
        xb, DD, w1, DD, nullptr, 0, hbuf, DFFN, b1, DFFN, DD);
    gemm_bf16_kernel<true, false, true, false, false><<<dim3(ROWS / 64, DD / 64), 128, 0, stream>>>(
        hbuf, DFFN, w2, DFFN, y, DD, nullptr, 0, b2, DD, DFFN);

    // --- residual + LN2 ---
    ln_kernel<<<ROWS, 256, 0, stream>>>(x, y, xb, ln2_g, ln2_b);

    // --- output heads: even rows -> out_l, odd rows -> out_r (lda=1024 strides rows) ---
    float* outf = (float*)d_out;
    gemm_bf16_kernel<true, false, true, false, false><<<dim3(HROWS / 64, HPAD / 64), 128, 0, stream>>>(
        xb, 2 * DD, wol, DD, outf, HDIM, nullptr, 0, b_out_l, HDIM, DD);
    gemm_bf16_kernel<true, false, true, false, false><<<dim3(HROWS / 64, HPAD / 64), 128, 0, stream>>>(
        xb + DD, 2 * DD, wor, DD, outf + (size_t)HROWS * HDIM, HDIM, nullptr, 0, b_out_r, HDIM, DD);

    // --- final masking ---
    mask_out_kernel<<<cgrid(2L * HROWS * HDIM), 256, 0, stream>>>(outf, mask_l, mask_r, fpm);
}
